// LRU_6416681141103
// MI455X (gfx1250) — compile-verified, matrix operand order checked by reading, not run on hardware
//
#include <hip/hip_runtime.h>
#include <hip/hip_bf16.h>

// ---------------- problem dims ----------------
#define L_DIM   8192
#define H_DIM   1024
#define N_DIM   1024
#define CHUNK   128
#define NCHUNK  (L_DIM / CHUNK)   // 64

// K staged per TDM load (elements), per bf16 plane
#define KSTAGE      128
// padded LDS bytes per staged plane: 16 rows * 128 elem * 2B = 4096B stream,
// TDM pad adds 16B per 1024B streamed -> 4096 + 64 = 4160
#define STG_PLANE   4160
#define STG_BUF     (4 * STG_PLANE)   // 4 planes (hi/lo x re/im) = 16640 B

typedef __attribute__((ext_vector_type(16))) __bf16 v16bf;
typedef __attribute__((ext_vector_type(8)))  __bf16 v8bf;
typedef __attribute__((ext_vector_type(8)))  float  v8f;
typedef __attribute__((ext_vector_type(4)))  unsigned int v4u;
typedef __attribute__((ext_vector_type(8)))  int v8i;
typedef __attribute__((ext_vector_type(4)))  int v4i;

// ---------------- WMMA helper ----------------
__device__ __forceinline__ v8f wmma_bf16(v16bf a, v16bf b, v8f c) {
  return __builtin_amdgcn_wmma_f32_16x16x32_bf16(false, a, false, b, (short)0, c,
                                                 false, false);
}

// ---------------- Tensor Data Mover: 16 x KSTAGE bf16 tile -> LDS ----------------
// Builds the D# per CDNA5 ISA (2D tensor, groups 2/3 zero).
// data_size=1 (2B), pad_enable, pad_interval=7 (256 dwords = 1KB),
// pad_amount=3 (4 dwords = 16B) so staged rows don't alias LDS banks while
// keeping every 1KB block 16B-aligned for b128 fragment reads.
// This toolchain exposes the 6-arg builtin: (v4u, v8i, v4i, v4i, v8i, cpol).
__device__ __forceinline__ void tdm_load_tile(unsigned int lds_off,
                                              const void* gsrc) {
  unsigned long long ga = (unsigned long long)(uintptr_t)gsrc;
  v4u g0 = { 1u,                                     // count=1, user mode
             lds_off,                                // D#.lds_addr
             (unsigned int)ga,                       // global_addr[31:0]
             (unsigned int)(((ga >> 32) & 0x01FFFFFFu) | (2u << 30)) }; // [56:32]|type=2
  v8i g1 = {
    (int)((1u << 16) | (1u << 20) | (7u << 22) | (3u << 25)), // data_size=2B, pad on
    (int)(((unsigned)H_DIM & 0xFFFFu) << 16),                 // tensor_dim0 lo16
    (int)((((unsigned)H_DIM >> 16) & 0xFFFFu) |
          (((unsigned)N_DIM & 0xFFFFu) << 16)),               // dim0 hi | dim1 lo
    (int)((((unsigned)N_DIM >> 16) & 0xFFFFu) |
          ((unsigned)KSTAGE << 16)),                          // dim1 hi | tile_dim0
    (int)16,                                                  // tile_dim1 = 16 rows
    (int)(unsigned)H_DIM,                                     // dim0_stride lo32
    0, 0 };
  v4i gz4 = { 0, 0, 0, 0 };
  v8i gz8 = { 0, 0, 0, 0, 0, 0, 0, 0 };
  __builtin_amdgcn_tensor_load_to_lds(g0, g1, gz4, gz4, gz8, 0);
}

// ---------------- fragment loaders ----------------
// A-frag (16x32 bf16) from global, ISA 7.12.2 layout: lane=M, half-wave K ranges.
__device__ __forceinline__ v16bf load_frag(const __bf16* __restrict__ base,
                                           int row0, int k0, int ld) {
  const int lane = threadIdx.x & 31;
  const int r    = lane & 15;
  const int half = lane >> 4;
  const __bf16* p = base + (size_t)(row0 + r) * (size_t)ld + (size_t)(k0 + half * 8);
  v8bf lo = *reinterpret_cast<const v8bf*>(p);
  v8bf hi = *reinterpret_cast<const v8bf*>(p + 16);
  return __builtin_shufflevector(lo, hi, 0,1,2,3,4,5,6,7,8,9,10,11,12,13,14,15);
}

// B-frag from a TDM-staged (padded) LDS plane; kk in [0, KSTAGE), multiple of 32.
// Element byte offset E -> padded offset E + (E>>10)*16 (matches TDM pad).
__device__ __forceinline__ v16bf load_frag_lds(const char* plane, int kk) {
  const int lane = threadIdx.x & 31;
  const int r    = lane & 15;
  const int half = lane >> 4;
  unsigned E0 = (unsigned)(r * (KSTAGE * 2) + (kk + half * 8) * 2);
  unsigned E1 = E0 + 32;
  unsigned o0 = E0 + (E0 >> 10) * 16;
  unsigned o1 = E1 + (E1 >> 10) * 16;
  v8bf lo = *reinterpret_cast<const v8bf*>(plane + o0);
  v8bf hi = *reinterpret_cast<const v8bf*>(plane + o1);
  return __builtin_shufflevector(lo, hi, 0,1,2,3,4,5,6,7,8,9,10,11,12,13,14,15);
}

__device__ __forceinline__ float2 cmul(float2 a, float2 b) {
  return make_float2(a.x * b.x - a.y * b.y, a.x * b.y + a.y * b.x);
}

// ---------------- scalar precompute ----------------
__global__ void lru_precompute(const float* __restrict__ nu_log,
                               const float* __restrict__ theta_log,
                               const float* __restrict__ gamma_log,
                               float2* __restrict__ lambda,
                               float2* __restrict__ eg) {
  int n = blockIdx.x * blockDim.x + threadIdx.x;
  if (n >= N_DIM) return;
  float mod = expf(-expf(nu_log[n]));
  float ph  = expf(theta_log[n]);
  lambda[n] = make_float2(mod * cosf(ph), mod * sinf(ph));
  float g = gamma_log[n];                 // reference uses exp(i*gamma_log)
  eg[n] = make_float2(cosf(g), sinf(g));
}

// ---------------- fp32 -> bf16 hi/lo split ----------------
__global__ void lru_split_bf16(const float* __restrict__ src,
                               __bf16* __restrict__ hi, __bf16* __restrict__ lo,
                               int count) {
  int i = blockIdx.x * blockDim.x + threadIdx.x;
  if (i >= count) return;
  float x = src[i];
  __bf16 h = (__bf16)x;
  hi[i] = h;
  lo[i] = (__bf16)(x - (float)h);
}

// ---------------- GEMM 1: Bu = exp_gamma[n] * (u @ B^T), complex ----------------
// Workgroup: 256 rows x 16 cols. 8 waves x (2 m-subtiles). B staged by TDM,
// double-buffered; 24 WMMAs per wave per K=32 step.
__global__ void __launch_bounds__(256)
lru_gemm_bu(const __bf16* __restrict__ urh, const __bf16* __restrict__ url,
            const __bf16* __restrict__ uih, const __bf16* __restrict__ uil,
            const __bf16* __restrict__ brh, const __bf16* __restrict__ brl,
            const __bf16* __restrict__ bih, const __bf16* __restrict__ bil,
            const float2* __restrict__ eg,
            float* __restrict__ bu_r, float* __restrict__ bu_i) {
  __shared__ __align__(16) char smem[2 * STG_BUF];
  const int tn  = (blockIdx.x & (N_DIM / 16 - 1)) * 16;
  const int tmB = (blockIdx.x / (N_DIM / 16)) * 256;
  const int tm0 = tmB + (threadIdx.x >> 5) * 32;
  const __bf16* planes[4] = { brh, brl, bih, bil };

  if (threadIdx.x < 32) {                       // wave 0 drives the TDM
#pragma unroll
    for (int p = 0; p < 4; ++p)
      tdm_load_tile((unsigned int)(uintptr_t)&smem[p * STG_PLANE],
                    planes[p] + (size_t)tn * H_DIM);
  }

  v8f aRR0 = {}, aRR1 = {}, aII0 = {}, aII1 = {}, aGI0 = {}, aGI1 = {};
  const int NSTG = H_DIM / KSTAGE;              // 8
  for (int s = 0; s < NSTG; ++s) {
    __syncthreads();                            // prior compute on next buffer done
    if (threadIdx.x < 32) {
      if (s + 1 < NSTG) {
        char* dst = &smem[((s + 1) & 1) * STG_BUF];
#pragma unroll
        for (int p = 0; p < 4; ++p)
          tdm_load_tile((unsigned int)(uintptr_t)(dst + p * STG_PLANE),
                        planes[p] + (size_t)tn * H_DIM + (size_t)(s + 1) * KSTAGE);
        __builtin_amdgcn_s_wait_tensorcnt(4);   // in-order => stage s complete
      } else {
        __builtin_amdgcn_s_wait_tensorcnt(0);
      }
    }
    __syncthreads();                            // staged B visible to all waves
    const char* buf = &smem[(s & 1) * STG_BUF];
#pragma unroll
    for (int kk = 0; kk < KSTAGE; kk += 32) {
      const int k = s * KSTAGE + kk;
      v16bf aRh0 = load_frag(urh, tm0,      k, H_DIM);
      v16bf aRl0 = load_frag(url, tm0,      k, H_DIM);
      v16bf aIh0 = load_frag(uih, tm0,      k, H_DIM);
      v16bf aIl0 = load_frag(uil, tm0,      k, H_DIM);
      v16bf aRh1 = load_frag(urh, tm0 + 16, k, H_DIM);
      v16bf aRl1 = load_frag(url, tm0 + 16, k, H_DIM);
      v16bf aIh1 = load_frag(uih, tm0 + 16, k, H_DIM);
      v16bf aIl1 = load_frag(uil, tm0 + 16, k, H_DIM);
      v16bf bRh = load_frag_lds(buf + 0 * STG_PLANE, kk);
      v16bf bRl = load_frag_lds(buf + 1 * STG_PLANE, kk);
      v16bf bIh = load_frag_lds(buf + 2 * STG_PLANE, kk);
      v16bf bIl = load_frag_lds(buf + 3 * STG_PLANE, kk);

      aRR0 = wmma_bf16(aRh0, bRh, aRR0);   // u_r*B_r (hi*hi+lo*hi+hi*lo)
      aRR0 = wmma_bf16(aRl0, bRh, aRR0);
      aRR0 = wmma_bf16(aRh0, bRl, aRR0);
      aII0 = wmma_bf16(aIh0, bIh, aII0);   // u_i*B_i
      aII0 = wmma_bf16(aIl0, bIh, aII0);
      aII0 = wmma_bf16(aIh0, bIl, aII0);
      aGI0 = wmma_bf16(aRh0, bIh, aGI0);   // u_r*B_i + u_i*B_r
      aGI0 = wmma_bf16(aRl0, bIh, aGI0);
      aGI0 = wmma_bf16(aRh0, bIl, aGI0);
      aGI0 = wmma_bf16(aIh0, bRh, aGI0);
      aGI0 = wmma_bf16(aIl0, bRh, aGI0);
      aGI0 = wmma_bf16(aIh0, bRl, aGI0);

      aRR1 = wmma_bf16(aRh1, bRh, aRR1);
      aRR1 = wmma_bf16(aRl1, bRh, aRR1);
      aRR1 = wmma_bf16(aRh1, bRl, aRR1);
      aII1 = wmma_bf16(aIh1, bIh, aII1);
      aII1 = wmma_bf16(aIl1, bIh, aII1);
      aII1 = wmma_bf16(aIh1, bIl, aII1);
      aGI1 = wmma_bf16(aRh1, bIh, aGI1);
      aGI1 = wmma_bf16(aRl1, bIh, aGI1);
      aGI1 = wmma_bf16(aRh1, bIl, aGI1);
      aGI1 = wmma_bf16(aIh1, bRh, aGI1);
      aGI1 = wmma_bf16(aIl1, bRh, aGI1);
      aGI1 = wmma_bf16(aIh1, bRl, aGI1);
    }
  }

  const int lane = threadIdx.x & 31;
  const int nn = lane & 15;
  const int mh = (lane >> 4) * 8;
  const float2 g = eg[tn + nn];
#pragma unroll
  for (int r = 0; r < 8; ++r) {
    float gr = aRR0[r] - aII0[r], gi = aGI0[r];
    size_t idx = (size_t)(tm0 + mh + r) * N_DIM + (size_t)(tn + nn);
    bu_r[idx] = gr * g.x - gi * g.y;
    bu_i[idx] = gr * g.y + gi * g.x;
    gr = aRR1[r] - aII1[r]; gi = aGI1[r];
    idx = (size_t)(tm0 + 16 + mh + r) * N_DIM + (size_t)(tn + nn);
    bu_r[idx] = gr * g.x - gi * g.y;
    bu_i[idx] = gr * g.y + gi * g.x;
  }
}

// ---------------- blocked linear scan: x[l] = Lambda*x[l-1] + Bu[l] ----------------
__global__ void lru_scan_local(const float2* __restrict__ lambda,
                               float* __restrict__ xr, float* __restrict__ xi,
                               float2* __restrict__ carry) {
  const int n = blockIdx.x * blockDim.x + threadIdx.x;
  const int c = blockIdx.y;
  const float2 lam = lambda[n];
  float2 s = make_float2(0.f, 0.f);
  size_t idx = (size_t)c * CHUNK * N_DIM + (size_t)n;
  for (int i = 0; i < CHUNK; ++i) {
    float br = xr[idx], bi = xi[idx];
    float sr = lam.x * s.x - lam.y * s.y + br;
    float si = lam.x * s.y + lam.y * s.x + bi;
    s = make_float2(sr, si);
    xr[idx] = sr; xi[idx] = si;
    idx += N_DIM;
  }
  carry[(size_t)c * N_DIM + n] = s;
}

__global__ void lru_scan_carry(const float2* __restrict__ lambda,
                               const float2* __restrict__ carry,
                               float2* __restrict__ prefix) {
  const int n = blockIdx.x * blockDim.x + threadIdx.x;
  float2 lamC = lambda[n];
  for (int i = 0; i < 7; ++i) lamC = cmul(lamC, lamC);   // Lambda^128
  float2 s = make_float2(0.f, 0.f);
  for (int c = 0; c < NCHUNK; ++c) {
    prefix[(size_t)c * N_DIM + n] = s;
    float2 cr = carry[(size_t)c * N_DIM + n];
    float2 t = cmul(lamC, s);
    s = make_float2(t.x + cr.x, t.y + cr.y);
  }
}

__global__ void lru_scan_fix(const float2* __restrict__ lambda,
                             const float2* __restrict__ prefix,
                             float* __restrict__ xr, float* __restrict__ xi) {
  const int n = blockIdx.x * blockDim.x + threadIdx.x;
  const int c = blockIdx.y;
  const float2 lam = lambda[n];
  float2 t = cmul(lam, prefix[(size_t)c * N_DIM + n]);
  size_t idx = (size_t)c * CHUNK * N_DIM + (size_t)n;
  for (int i = 0; i < CHUNK; ++i) {
    xr[idx] += t.x;
    xi[idx] += t.y;
    t = cmul(lam, t);
    idx += N_DIM;
  }
}

// ---------------- GEMM 2: y = Re(x @ C^T) + D*u_r ----------------
// Same macro-tile/TDM structure; 12 WMMAs per wave per K=32 step.
__global__ void __launch_bounds__(256)
lru_gemm_y(const __bf16* __restrict__ xrh, const __bf16* __restrict__ xrl,
           const __bf16* __restrict__ xih, const __bf16* __restrict__ xil,
           const __bf16* __restrict__ crh, const __bf16* __restrict__ crl,
           const __bf16* __restrict__ cih, const __bf16* __restrict__ cil,
           const float* __restrict__ Dv, const float* __restrict__ in_real,
           float* __restrict__ out) {
  __shared__ __align__(16) char smem[2 * STG_BUF];
  const int th  = (blockIdx.x & (H_DIM / 16 - 1)) * 16;
  const int tmB = (blockIdx.x / (H_DIM / 16)) * 256;
  const int tm0 = tmB + (threadIdx.x >> 5) * 32;
  const __bf16* planes[4] = { crh, crl, cih, cil };

  if (threadIdx.x < 32) {
#pragma unroll
    for (int p = 0; p < 4; ++p)
      tdm_load_tile((unsigned int)(uintptr_t)&smem[p * STG_PLANE],
                    planes[p] + (size_t)th * N_DIM);
  }

  v8f aR0 = {}, aR1 = {}, aI0 = {}, aI1 = {};
  const int NSTG = N_DIM / KSTAGE;
  for (int s = 0; s < NSTG; ++s) {
    __syncthreads();
    if (threadIdx.x < 32) {
      if (s + 1 < NSTG) {
        char* dst = &smem[((s + 1) & 1) * STG_BUF];
#pragma unroll
        for (int p = 0; p < 4; ++p)
          tdm_load_tile((unsigned int)(uintptr_t)(dst + p * STG_PLANE),
                        planes[p] + (size_t)th * N_DIM + (size_t)(s + 1) * KSTAGE);
        __builtin_amdgcn_s_wait_tensorcnt(4);
      } else {
        __builtin_amdgcn_s_wait_tensorcnt(0);
      }
    }
    __syncthreads();
    const char* buf = &smem[(s & 1) * STG_BUF];
#pragma unroll
    for (int kk = 0; kk < KSTAGE; kk += 32) {
      const int k = s * KSTAGE + kk;
      v16bf aRh0 = load_frag(xrh, tm0,      k, N_DIM);
      v16bf aRl0 = load_frag(xrl, tm0,      k, N_DIM);
      v16bf aIh0 = load_frag(xih, tm0,      k, N_DIM);
      v16bf aIl0 = load_frag(xil, tm0,      k, N_DIM);
      v16bf aRh1 = load_frag(xrh, tm0 + 16, k, N_DIM);
      v16bf aRl1 = load_frag(xrl, tm0 + 16, k, N_DIM);
      v16bf aIh1 = load_frag(xih, tm0 + 16, k, N_DIM);
      v16bf aIl1 = load_frag(xil, tm0 + 16, k, N_DIM);
      v16bf bRh = load_frag_lds(buf + 0 * STG_PLANE, kk);
      v16bf bRl = load_frag_lds(buf + 1 * STG_PLANE, kk);
      v16bf bIh = load_frag_lds(buf + 2 * STG_PLANE, kk);
      v16bf bIl = load_frag_lds(buf + 3 * STG_PLANE, kk);

      aR0 = wmma_bf16(aRh0, bRh, aR0);   // x_r*C_r
      aR0 = wmma_bf16(aRl0, bRh, aR0);
      aR0 = wmma_bf16(aRh0, bRl, aR0);
      aI0 = wmma_bf16(aIh0, bIh, aI0);   // x_i*C_i
      aI0 = wmma_bf16(aIl0, bIh, aI0);
      aI0 = wmma_bf16(aIh0, bIl, aI0);

      aR1 = wmma_bf16(aRh1, bRh, aR1);
      aR1 = wmma_bf16(aRl1, bRh, aR1);
      aR1 = wmma_bf16(aRh1, bRl, aR1);
      aI1 = wmma_bf16(aIh1, bIh, aI1);
      aI1 = wmma_bf16(aIl1, bIh, aI1);
      aI1 = wmma_bf16(aIh1, bIl, aI1);
    }
  }

  const int lane = threadIdx.x & 31;
  const int nn = lane & 15;
  const int mh = (lane >> 4) * 8;
  const int col = th + nn;
  const float d = Dv[col];
#pragma unroll
  for (int r = 0; r < 8; ++r) {
    size_t idx = (size_t)(tm0 + mh + r) * H_DIM + (size_t)col;
    out[idx] = (aR0[r] - aI0[r]) + d * in_real[idx];
    idx = (size_t)(tm0 + 16 + mh + r) * H_DIM + (size_t)col;
    out[idx] = (aR1[r] - aI1[r]) + d * in_real[idx];
  }
}

// ---------------- launcher ----------------
extern "C" void kernel_launch(void* const* d_in, const int* in_sizes, int n_in,
                              void* d_out, int out_size, void* d_ws, size_t ws_size,
                              hipStream_t stream) {
  const float* in_real  = (const float*)d_in[0];
  const float* in_imag  = (const float*)d_in[1];
  const float* nu_log   = (const float*)d_in[2];
  const float* theta_log= (const float*)d_in[3];
  const float* B_real   = (const float*)d_in[4];
  const float* B_imag   = (const float*)d_in[5];
  const float* C_real   = (const float*)d_in[6];
  const float* C_imag   = (const float*)d_in[7];
  const float* Dv       = (const float*)d_in[8];
  const float* gamma_log= (const float*)d_in[9];
  float* out = (float*)d_out;

  const size_t LH = (size_t)L_DIM * H_DIM;
  const size_t NH = (size_t)N_DIM * H_DIM;
  const size_t LN = (size_t)L_DIM * N_DIM;

  char* w = (char*)d_ws;
  size_t off = 0;
  auto take = [&](size_t bytes) -> void* {
    void* p = w + off;
    off += (bytes + 255) & ~(size_t)255;
    return p;
  };
  __bf16* ur_hi = (__bf16*)take(LH * 2);   // later reused as x planes
  __bf16* ur_lo = (__bf16*)take(LH * 2);
  __bf16* ui_hi = (__bf16*)take(LH * 2);
  __bf16* ui_lo = (__bf16*)take(LH * 2);
  __bf16* br_hi = (__bf16*)take(NH * 2);   // later reused as C planes
  __bf16* br_lo = (__bf16*)take(NH * 2);
  __bf16* bi_hi = (__bf16*)take(NH * 2);
  __bf16* bi_lo = (__bf16*)take(NH * 2);
  float*  bu_r  = (float*)take(LN * 4);    // scanned in place -> x
  float*  bu_i  = (float*)take(LN * 4);
  float2* lambda= (float2*)take(N_DIM * sizeof(float2));
  float2* eg    = (float2*)take(N_DIM * sizeof(float2));
  float2* carry = (float2*)take((size_t)NCHUNK * N_DIM * sizeof(float2));
  float2* prefix= (float2*)take((size_t)NCHUNK * N_DIM * sizeof(float2));
  (void)ws_size; (void)in_sizes; (void)n_in; (void)out_size;

  lru_precompute<<<dim3(N_DIM / 256), dim3(256), 0, stream>>>(nu_log, theta_log,
                                                              gamma_log, lambda, eg);
  lru_split_bf16<<<dim3((unsigned)(LH / 256)), dim3(256), 0, stream>>>(in_real, ur_hi, ur_lo, (int)LH);
  lru_split_bf16<<<dim3((unsigned)(LH / 256)), dim3(256), 0, stream>>>(in_imag, ui_hi, ui_lo, (int)LH);
  lru_split_bf16<<<dim3((unsigned)(NH / 256)), dim3(256), 0, stream>>>(B_real, br_hi, br_lo, (int)NH);
  lru_split_bf16<<<dim3((unsigned)(NH / 256)), dim3(256), 0, stream>>>(B_imag, bi_hi, bi_lo, (int)NH);

  {  // (L/256) * (N/16) workgroups
    unsigned blocks = (L_DIM / 256) * (N_DIM / 16);
    lru_gemm_bu<<<dim3(blocks), dim3(256), 0, stream>>>(
        ur_hi, ur_lo, ui_hi, ui_lo, br_hi, br_lo, bi_hi, bi_lo, eg, bu_r, bu_i);
  }

  lru_scan_local<<<dim3(N_DIM / 256, NCHUNK), dim3(256), 0, stream>>>(lambda, bu_r, bu_i, carry);
  lru_scan_carry<<<dim3(N_DIM / 256), dim3(256), 0, stream>>>(lambda, carry, prefix);
  lru_scan_fix<<<dim3(N_DIM / 256, NCHUNK), dim3(256), 0, stream>>>(lambda, prefix, bu_r, bu_i);

  lru_split_bf16<<<dim3((unsigned)(LN / 256)), dim3(256), 0, stream>>>(bu_r, ur_hi, ur_lo, (int)LN);
  lru_split_bf16<<<dim3((unsigned)(LN / 256)), dim3(256), 0, stream>>>(bu_i, ui_hi, ui_lo, (int)LN);
  lru_split_bf16<<<dim3((unsigned)(NH / 256)), dim3(256), 0, stream>>>(C_real, br_hi, br_lo, (int)NH);
  lru_split_bf16<<<dim3((unsigned)(NH / 256)), dim3(256), 0, stream>>>(C_imag, bi_hi, bi_lo, (int)NH);

  {  // (L/256) * (H/16) workgroups
    unsigned blocks = (L_DIM / 256) * (H_DIM / 16);
    lru_gemm_y<<<dim3(blocks), dim3(256), 0, stream>>>(
        ur_hi, ur_lo, ui_hi, ui_lo, br_hi, br_lo, bi_hi, bi_lo, Dv, in_real, out);
  }
}
